// AdvancedLSTMCell_30270929502695
// MI455X (gfx1250) — matrix operand order repaired and checked by reading, not compiled, not run on hardware
//
#include <hip/hip_runtime.h>
#include <hip/hip_bf16.h>
#include <stdint.h>

// ---------------------------------------------------------------------------
// Problem dimensions (AdvancedLSTMCell): B=16384, IN=512, H=1024
// ---------------------------------------------------------------------------
#define BDIM   16384
#define INDIM  512
#define HDIM   1024
#define KDIM   (INDIM + HDIM)   // 1536

typedef __attribute__((ext_vector_type(16))) __bf16 v16bf;
typedef __attribute__((ext_vector_type(8)))  float  v8f;

union Frag16 { v16bf v; uint4 q[2]; };

__device__ __forceinline__ unsigned short f32_to_bf16(float f) {
    union { float f; uint32_t u; } v; v.f = f;
    uint32_t u = v.u;
    uint32_t r = u + 0x7FFFu + ((u >> 16) & 1u);     // round-to-nearest-even
    if ((u & 0x7F800000u) == 0x7F800000u) r = u;     // inf/nan passthrough
    return (unsigned short)(r >> 16);
}

__device__ __forceinline__ float sigf(float x) { return 1.0f / (1.0f + __expf(-x)); }

// ---------------------------------------------------------------------------
// CDNA5 async global->LDS copy (ASYNCcnt-tracked, bypasses VGPR staging).
//   vDst = VGPR holding LDS byte address, vAddr = 64-bit global address.
// ---------------------------------------------------------------------------
__device__ __forceinline__ void async_ld16(unsigned lds_addr, const void* gptr) {
    asm volatile("global_load_async_to_lds_b128 %0, %1, off"
                 :: "v"(lds_addr), "v"((uint64_t)(uintptr_t)gptr)
                 : "memory");
}
__device__ __forceinline__ void wait_async0() {
    asm volatile("s_wait_asynccnt 0" ::: "memory");
}

// ---------------------------------------------------------------------------
// Pack kernels: fp32 -> bf16 staging buffers
// ---------------------------------------------------------------------------
__global__ __launch_bounds__(256) void k_pack_combined(
    const float* __restrict__ x, const float* __restrict__ h,
    unsigned short* __restrict__ out)
{
    int64_t i = (int64_t)blockIdx.x * 256 + threadIdx.x;   // over B*KDIM
    int b = (int)(i / KDIM), c = (int)(i % KDIM);
    float v = (c < INDIM) ? x[(int64_t)b * INDIM + c]
                          : h[(int64_t)b * HDIM + (c - INDIM)];
    out[i] = f32_to_bf16(v);
}

__global__ __launch_bounds__(256) void k_pack_wgates(
    const float* __restrict__ Wi, const float* __restrict__ Wf,
    const float* __restrict__ Wo, const float* __restrict__ Wg,
    unsigned short* __restrict__ out)            // [4*H][KDIM]
{
    int64_t i = (int64_t)blockIdx.x * 256 + threadIdx.x;   // over 4*H*KDIM
    int g = (int)(i / ((int64_t)HDIM * KDIM));
    int64_t r = i % ((int64_t)HDIM * KDIM);
    const float* src = (g == 0) ? Wi : (g == 1) ? Wf : (g == 2) ? Wo : Wg;
    out[i] = f32_to_bf16(src[r]);
}

__global__ __launch_bounds__(256) void k_pack_whw(
    const float* __restrict__ Wht, const float* __restrict__ Whg,
    unsigned short* __restrict__ out)            // [2*H][INDIM]
{
    int64_t i = (int64_t)blockIdx.x * 256 + threadIdx.x;   // over 2*H*INDIM
    const int64_t half = (int64_t)HDIM * INDIM;
    float v = (i < half) ? Wht[i] : Whg[i - half];
    out[i] = f32_to_bf16(v);
}

// ---------------------------------------------------------------------------
// Fused gate GEMM + LSTM cell update.
//   gates = combined[B,1536] @ W_gates^T[1536,4096] (+bias)
//   Block tile: 128 rows x 64 H-columns, all 4 gate groups; 8 waves, each
//   wave 32x32 per gate group. Double-buffered LDS, async global->LDS.
// ---------------------------------------------------------------------------
#define LDA 40   // 32 bf16 + 8 pad (16B) per LDS row

__global__ __launch_bounds__(256) void k_gates_gemm(
    const unsigned short* __restrict__ A,    // combined bf16 [B][1536]
    const unsigned short* __restrict__ W,    // W_gates bf16 [4096][1536]
    const float* __restrict__ bi, const float* __restrict__ bf,
    const float* __restrict__ bo, const float* __restrict__ bg,
    const float* __restrict__ c_prev,
    float* __restrict__ c_out,               // -> d_out + B*H
    float* __restrict__ h_pre)               // -> ws (pre-attention h)
{
    __shared__ __align__(16) unsigned short lA[2][128][LDA];
    __shared__ __align__(16) unsigned short lB[2][4][64][LDA];

    const int tid  = threadIdx.x;
    const int lane = tid & 31;
    const int wid  = tid >> 5;
    const int wr = wid >> 1, wc = wid & 1;      // wave grid 4x2
    const int rowbase = blockIdx.x * 128;
    const int colbase = blockIdx.y * 64;        // within H

    v8f acc[4][2][2];
    #pragma unroll
    for (int g = 0; g < 4; ++g)
        #pragma unroll
        for (int mi = 0; mi < 2; ++mi)
            #pragma unroll
            for (int ni = 0; ni < 2; ++ni)
                acc[g][mi][ni] = (v8f){0.f,0.f,0.f,0.f,0.f,0.f,0.f,0.f};

    const int am  = lane & 15;
    const int akb = (lane >> 4) * 8;            // A frag K-chunk base
    const int bn  = lane & 15;
    const int bkb = (lane >> 4) * 8;            // B frag K base

    // issue async staging of one K-slab (A: 2x16B, B: 4x16B per thread)
    auto issue_tile = [&](int buf, int kt) {
        #pragma unroll
        for (int i = 0; i < 2; ++i) {
            int idx = tid + i * 256;
            int r = idx >> 2, s = idx & 3;
            async_ld16((unsigned)(uintptr_t)&lA[buf][r][s * 8],
                       &A[(int64_t)(rowbase + r) * KDIM + kt + s * 8]);
        }
        #pragma unroll
        for (int i = 0; i < 4; ++i) {
            int idx = tid + i * 256;
            int g = idx >> 8, rem = idx & 255;
            int r = rem >> 2, s = rem & 3;
            async_ld16((unsigned)(uintptr_t)&lB[buf][g][r][s * 8],
                       &W[(int64_t)(g * HDIM + colbase + r) * KDIM + kt + s * 8]);
        }
    };

    const int nk = KDIM / 32;
    int buf = 0;
    issue_tile(0, 0);
    wait_async0();
    __syncthreads();

    for (int it = 0; it < nk; ++it) {
        if (it + 1 < nk) issue_tile(buf ^ 1, (it + 1) * 32);   // overlap with WMMA

        // A fragments for this wave
        Frag16 af[2];
        #pragma unroll
        for (int mi = 0; mi < 2; ++mi) {
            int row = wr * 32 + mi * 16 + am;
            af[mi].q[0] = *(const uint4*)&lA[buf][row][akb];
            af[mi].q[1] = *(const uint4*)&lA[buf][row][akb + 16];
        }
        // B fragments: 2-deep software pipeline across the 4 gate groups
        Frag16 bq0[2], bq1[2];
        #pragma unroll
        for (int ni = 0; ni < 2; ++ni) {
            int col = wc * 32 + ni * 16 + bn;
            bq0[ni].q[0] = *(const uint4*)&lB[buf][0][col][bkb];
            bq0[ni].q[1] = *(const uint4*)&lB[buf][0][col][bkb + 16];
        }
        #pragma unroll
        for (int g = 0; g < 4; ++g) {
            Frag16* cur = (g & 1) ? bq1 : bq0;
            Frag16* nxt = (g & 1) ? bq0 : bq1;
            if (g < 3) {
                #pragma unroll
                for (int ni = 0; ni < 2; ++ni) {
                    int col = wc * 32 + ni * 16 + bn;
                    nxt[ni].q[0] = *(const uint4*)&lB[buf][g + 1][col][bkb];
                    nxt[ni].q[1] = *(const uint4*)&lB[buf][g + 1][col][bkb + 16];
                }
            }
            #pragma unroll
            for (int ni = 0; ni < 2; ++ni)
                #pragma unroll
                for (int mi = 0; mi < 2; ++mi)
                    acc[g][mi][ni] = __builtin_amdgcn_wmma_f32_16x16x32_bf16(
                        false, af[mi].v, false, cur[ni].v, (short)0,
                        acc[g][mi][ni], false, false);
        }

        wait_async0();        // next tile landed in other buffer
        __syncthreads();      // all lanes' async + LDS reads done
        buf ^= 1;
    }

    // --- fused LSTM epilogue: C/D layout lane->(n = lane&15, m = 8*(lane>=16)+r)
    const int n_l   = lane & 15;
    const int m_off = (lane >> 4) * 8;
    #pragma unroll
    for (int mi = 0; mi < 2; ++mi) {
        #pragma unroll
        for (int ni = 0; ni < 2; ++ni) {
            int col = colbase + wc * 32 + ni * 16 + n_l;
            float bbi = bi[col], bbf = bf[col], bbo = bo[col], bbg = bg[col];
            #pragma unroll
            for (int r = 0; r < 8; ++r) {
                int row = rowbase + wr * 32 + mi * 16 + m_off + r;
                float iv = sigf(acc[0][mi][ni][r] + bbi);
                float fv = sigf(acc[1][mi][ni][r] + bbf);
                float ov = sigf(acc[2][mi][ni][r] + bbo);
                float gv = tanhf(acc[3][mi][ni][r] + bbg);
                float cv = fv * c_prev[(int64_t)row * HDIM + col] + iv * gv;
                c_out[(int64_t)row * HDIM + col] = cv;
                h_pre[(int64_t)row * HDIM + col] = ov * tanhf(cv);
            }
        }
    }
}

// ---------------------------------------------------------------------------
// Attention logits: logits[b] = h_pre[b,:] . W_a + b_a   (wave per row)
// ---------------------------------------------------------------------------
__global__ __launch_bounds__(256) void k_logits(
    const float* __restrict__ h_pre, const float* __restrict__ Wa,
    const float* __restrict__ ba, float* __restrict__ logits)
{
    int row  = blockIdx.x * 8 + (threadIdx.x >> 5);
    int lane = threadIdx.x & 31;
    float s = 0.f;
    for (int k = lane; k < HDIM; k += 32)
        s += h_pre[(int64_t)row * HDIM + k] * Wa[k];
    #pragma unroll
    for (int off = 16; off > 0; off >>= 1) s += __shfl_down(s, off, 32);
    if (lane == 0) logits[row] = s + ba[0];
}

// ---------------------------------------------------------------------------
// Batch softmax stats over 16384 logits: stats = {max, 1/sum(exp(l-max))}
// ---------------------------------------------------------------------------
__global__ __launch_bounds__(1024) void k_softmax_stats(
    const float* __restrict__ logits, float* __restrict__ stats)
{
    __shared__ float red[1024];
    int t = threadIdx.x;
    float m = -INFINITY;
    for (int i = t; i < BDIM; i += 1024) m = fmaxf(m, logits[i]);
    red[t] = m; __syncthreads();
    for (int s = 512; s > 0; s >>= 1) {
        if (t < s) red[t] = fmaxf(red[t], red[t + s]);
        __syncthreads();
    }
    float mx = red[0]; __syncthreads();
    float sum = 0.f;
    for (int i = t; i < BDIM; i += 1024) sum += __expf(logits[i] - mx);
    red[t] = sum; __syncthreads();
    for (int s = 512; s > 0; s >>= 1) {
        if (t < s) red[t] += red[t + s];
        __syncthreads();
    }
    if (t == 0) { stats[0] = mx; stats[1] = 1.0f / red[0]; }
}

// ---------------------------------------------------------------------------
// Fused highway GEMM + final merge.
//   [t|gate_pre] = x[B,512] @ W_hw^T[512,2048] (+bias); A = combined (lda=1536)
//   h = sigmoid(gate_pre)*t + (1-sigmoid)*(h_pre * attn)
// ---------------------------------------------------------------------------
__global__ __launch_bounds__(256) void k_highway_gemm(
    const unsigned short* __restrict__ A,     // combined bf16, use K=0..511
    const unsigned short* __restrict__ W,     // W_hw bf16 [2048][512]
    const float* __restrict__ bht, const float* __restrict__ bhg,
    const float* __restrict__ h_pre, const float* __restrict__ logits,
    const float* __restrict__ stats,
    float* __restrict__ h_out)                // -> d_out
{
    __shared__ __align__(16) unsigned short lA[2][128][LDA];
    __shared__ __align__(16) unsigned short lB[2][2][64][LDA];

    const int tid  = threadIdx.x;
    const int lane = tid & 31;
    const int wid  = tid >> 5;
    const int wr = wid >> 1, wc = wid & 1;
    const int rowbase = blockIdx.x * 128;
    const int colbase = blockIdx.y * 64;

    v8f acc[2][2][2];
    #pragma unroll
    for (int g = 0; g < 2; ++g)
        #pragma unroll
        for (int mi = 0; mi < 2; ++mi)
            #pragma unroll
            for (int ni = 0; ni < 2; ++ni)
                acc[g][mi][ni] = (v8f){0.f,0.f,0.f,0.f,0.f,0.f,0.f,0.f};

    const int am  = lane & 15;
    const int akb = (lane >> 4) * 8;
    const int bn  = lane & 15;
    const int bkb = (lane >> 4) * 8;

    auto issue_tile = [&](int buf, int kt) {
        #pragma unroll
        for (int i = 0; i < 2; ++i) {
            int idx = tid + i * 256;
            int r = idx >> 2, s = idx & 3;
            async_ld16((unsigned)(uintptr_t)&lA[buf][r][s * 8],
                       &A[(int64_t)(rowbase + r) * KDIM + kt + s * 8]);
        }
        #pragma unroll
        for (int i = 0; i < 2; ++i) {
            int idx = tid + i * 256;
            int g = idx >> 8, rem = idx & 255;
            int r = rem >> 2, s = rem & 3;
            async_ld16((unsigned)(uintptr_t)&lB[buf][g][r][s * 8],
                       &W[(int64_t)(g * HDIM + colbase + r) * INDIM + kt + s * 8]);
        }
    };

    const int nk = INDIM / 32;
    int buf = 0;
    issue_tile(0, 0);
    wait_async0();
    __syncthreads();

    for (int it = 0; it < nk; ++it) {
        if (it + 1 < nk) issue_tile(buf ^ 1, (it + 1) * 32);

        Frag16 af[2];
        #pragma unroll
        for (int mi = 0; mi < 2; ++mi) {
            int row = wr * 32 + mi * 16 + am;
            af[mi].q[0] = *(const uint4*)&lA[buf][row][akb];
            af[mi].q[1] = *(const uint4*)&lA[buf][row][akb + 16];
        }
        Frag16 bq[2][2];
        #pragma unroll
        for (int g = 0; g < 2; ++g)
            #pragma unroll
            for (int ni = 0; ni < 2; ++ni) {
                int col = wc * 32 + ni * 16 + bn;
                bq[g][ni].q[0] = *(const uint4*)&lB[buf][g][col][bkb];
                bq[g][ni].q[1] = *(const uint4*)&lB[buf][g][col][bkb + 16];
            }
        #pragma unroll
        for (int g = 0; g < 2; ++g)
            #pragma unroll
            for (int ni = 0; ni < 2; ++ni)
                #pragma unroll
                for (int mi = 0; mi < 2; ++mi)
                    acc[g][mi][ni] = __builtin_amdgcn_wmma_f32_16x16x32_bf16(
                        false, af[mi].v, false, bq[g][ni].v, (short)0,
                        acc[g][mi][ni], false, false);

        wait_async0();
        __syncthreads();
        buf ^= 1;
    }

    const float mx   = stats[0];
    const float isum = stats[1];
    const int n_l    = lane & 15;
    const int m_off  = (lane >> 4) * 8;
    #pragma unroll
    for (int mi = 0; mi < 2; ++mi) {
        #pragma unroll
        for (int ni = 0; ni < 2; ++ni) {
            int col = colbase + wc * 32 + ni * 16 + n_l;
            float bt = bht[col], bg2 = bhg[col];
            #pragma unroll
            for (int r = 0; r < 8; ++r) {
                int row = rowbase + wr * 32 + mi * 16 + m_off + r;
                float tv   = acc[0][mi][ni][r] + bt;
                float gate = sigf(acc[1][mi][ni][r] + bg2);
                float att  = __expf(logits[row] - mx) * isum;
                float ha   = h_pre[(int64_t)row * HDIM + col] * att;
                h_out[(int64_t)row * HDIM + col] = gate * tv + (1.0f - gate) * ha;
            }
        }
    }
}

// ---------------------------------------------------------------------------
// Launcher
// ---------------------------------------------------------------------------
extern "C" void kernel_launch(void* const* d_in, const int* in_sizes, int n_in,
                              void* d_out, int out_size, void* d_ws, size_t ws_size,
                              hipStream_t stream)
{
    (void)in_sizes; (void)n_in; (void)out_size; (void)ws_size;

    const float* x      = (const float*)d_in[0];
    const float* h_prev = (const float*)d_in[1];
    const float* c_prev = (const float*)d_in[2];
    const float* W_i = (const float*)d_in[3];  const float* b_i = (const float*)d_in[4];
    const float* W_f = (const float*)d_in[5];  const float* b_f = (const float*)d_in[6];
    const float* W_o = (const float*)d_in[7];  const float* b_o = (const float*)d_in[8];
    const float* W_g = (const float*)d_in[9];  const float* b_g = (const float*)d_in[10];
    const float* W_a = (const float*)d_in[11]; const float* b_a = (const float*)d_in[12];
    const float* W_ht = (const float*)d_in[13]; const float* b_ht = (const float*)d_in[14];
    const float* W_hg = (const float*)d_in[15]; const float* b_hg = (const float*)d_in[16];

    float* h_out = (float*)d_out;                          // [B*H]
    float* c_out = (float*)d_out + (int64_t)BDIM * HDIM;   // [B*H]

    // Workspace layout (16B-aligned slices)
    char* ws = (char*)d_ws;
    unsigned short* combined = (unsigned short*)ws;                          // B*1536 bf16
    ws += (int64_t)BDIM * KDIM * sizeof(unsigned short);                     // 48 MB
    unsigned short* Wg_bf = (unsigned short*)ws;                             // 4096*1536 bf16
    ws += (int64_t)4 * HDIM * KDIM * sizeof(unsigned short);                 // 12 MB
    unsigned short* Whw_bf = (unsigned short*)ws;                            // 2048*512 bf16
    ws += (int64_t)2 * HDIM * INDIM * sizeof(unsigned short);                // 2 MB
    float* h_pre = (float*)ws;                                               // B*H f32
    ws += (int64_t)BDIM * HDIM * sizeof(float);                              // 64 MB
    float* logits = (float*)ws;
    ws += (int64_t)BDIM * sizeof(float);                                     // 64 KB
    float* stats = (float*)ws;                                               // {max, 1/sum}

    // 1) pack activations + weights to bf16
    k_pack_combined<<<(BDIM * KDIM) / 256, 256, 0, stream>>>(x, h_prev, combined);
    k_pack_wgates<<<(4 * HDIM * KDIM) / 256, 256, 0, stream>>>(W_i, W_f, W_o, W_g, Wg_bf);
    k_pack_whw<<<(2 * HDIM * INDIM) / 256, 256, 0, stream>>>(W_ht, W_hg, Whw_bf);

    // 2) fused gate GEMM + LSTM cell update
    dim3 gg(BDIM / 128, HDIM / 64);
    k_gates_gemm<<<gg, 256, 0, stream>>>(combined, Wg_bf, b_i, b_f, b_o, b_g,
                                         c_prev, c_out, h_pre);

    // 3) attention logits + batch softmax stats
    k_logits<<<BDIM / 8, 256, 0, stream>>>(h_pre, W_a, b_a, logits);
    k_softmax_stats<<<1, 1024, 0, stream>>>(logits, stats);

    // 4) fused highway GEMM + attention-scaled merge
    k_highway_gemm<<<gg, 256, 0, stream>>>(combined, Whw_bf, b_ht, b_hg,
                                           h_pre, logits, stats, h_out);
}
